// FlashDiffAttention_82789789598491
// MI455X (gfx1250) — matrix operand order repaired and checked by reading, not compiled, not run on hardware
//
#include <hip/hip_runtime.h>
#include <hip/hip_fp16.h>

typedef __attribute__((ext_vector_type(16))) _Float16 v16h;
typedef __attribute__((ext_vector_type(8)))  _Float16 v8h;
typedef __attribute__((ext_vector_type(2)))  _Float16 v2h;
typedef __attribute__((ext_vector_type(8)))  float    v8f;

#define H2 16     // total heads
#define NH 8      // differential head pairs
#define HD 128    // head dim
#define BM 32     // query rows per block
#define BN 32     // keys per iteration

// LDS (bytes), loop phase:
//   k_lds : [2][32][128] f16 = 16384  (K for group0/group1 heads, row-major)
//   v_lds : [256][32]    f16 = 16384  (V transposed [vdim][key]; vdim>=128 -> v2)
//   p_lds : [4][16*32]   f16 =  4096  (per-wave P staging, C-layout -> A-layout)
// combine phase (after barrier, overlaps): cb [32][256] f32 = 32768
#define SMEM_BYTES 36864

// ---- 16-lane xor-butterfly stages.
// On gfx1250, v_permlane16_b32 permutes within 16-lane groups in one VALU op
// (vs ds_bpermute for __shfl_xor), keeping the DS pipe free for WMMA fragments.
#if __has_builtin(__builtin_amdgcn_permlane16)
template <unsigned LO, unsigned HI>
__device__ __forceinline__ float plx16(float x) {
  int v = __builtin_bit_cast(int, x);
  v = __builtin_amdgcn_permlane16(v, v, LO, HI, false, false);
  return __builtin_bit_cast(float, v);
}
__device__ __forceinline__ float bfly16_max(float t) {
  t = fmaxf(t, plx16<0x67452301u, 0xEFCDAB89u>(t));  // xor 1
  t = fmaxf(t, plx16<0x54761032u, 0xDCFE98BAu>(t));  // xor 2
  t = fmaxf(t, plx16<0x32107654u, 0xBA98FEDCu>(t));  // xor 4
  t = fmaxf(t, plx16<0xFEDCBA98u, 0x76543210u>(t));  // xor 8
  return t;
}
__device__ __forceinline__ float bfly16_sum(float t) {
  t += plx16<0x67452301u, 0xEFCDAB89u>(t);
  t += plx16<0x54761032u, 0xDCFE98BAu>(t);
  t += plx16<0x32107654u, 0xBA98FEDCu>(t);
  t += plx16<0xFEDCBA98u, 0x76543210u>(t);
  return t;
}
#else
__device__ __forceinline__ float bfly16_max(float t) {
  #pragma unroll
  for (int x = 1; x < 16; x <<= 1) t = fmaxf(t, __shfl_xor(t, x, 32));
  return t;
}
__device__ __forceinline__ float bfly16_sum(float t) {
  #pragma unroll
  for (int x = 1; x < 16; x <<= 1) t += __shfl_xor(t, x, 32);
  return t;
}
#endif

__global__ __launch_bounds__(128)
void FlashDiffAttention_82789789598491_kernel(
    const float* __restrict__ q, const float* __restrict__ k,
    const float* __restrict__ v,
    const float* __restrict__ lq1, const float* __restrict__ lk1,
    const float* __restrict__ lq2, const float* __restrict__ lk2,
    float* __restrict__ out, int S) {
  __shared__ char smem[SMEM_BYTES];
  __shared__ float s_lambda;
  _Float16* k_lds = (_Float16*)smem;             // [2][32][128]
  _Float16* v_lds = (_Float16*)(smem + 16384);   // [256][32]
  _Float16* p_lds = (_Float16*)(smem + 32768);   // [4][512]
  float*    cb    = (float*)smem;                // [32][256] (combine phase)

  const int tid  = threadIdx.x;
  const int lane = tid & 31;
  const int wid  = tid >> 5;      // 0..3
  const int grp  = wid >> 1;      // 0 = attn1 waves, 1 = attn2 waves
  const int slc  = wid & 1;       // 16-row slice within 32-row tile
  const int hl   = lane >> 4;     // lane half (M+8 / K+16 group select)
  const int ln   = lane & 15;

  const int mb = blockIdx.x * BM; // query tile base
  const int h  = blockIdx.y;      // head pair
  const int b  = blockIdx.z;

  const float LAM_INIT = 0.8f - 0.6f * __expf(-3.6f);

  if (wid == 0) {  // lambda_full: wave-parallel dot products over 128 dims
    float a1 = 0.f, a2 = 0.f;
    #pragma unroll
    for (int i = 0; i < 4; ++i) {
      int idx = lane + 32 * i;
      a1 += lq1[idx] * lk1[idx];
      a2 += lq2[idx] * lk2[idx];
    }
    a1 = bfly16_sum(a1); a1 += __shfl_xor(a1, 16, 32);
    a2 = bfly16_sum(a2); a2 += __shfl_xor(a2, 16, 32);
    if (lane == 0) s_lambda = __expf(a1) - __expf(a2) + LAM_INIT;
  }

  const int   m0    = mb + slc * 16;            // wave's first query row
  const float scale = 0.08838834764831845f;     // 1/sqrt(128)

  // ---- preload Q as 4 A-fragments (16x32 f16) covering D=128.
  // A layout: lane L: row = L&15; element i -> K = 32f + 16*hl + i (contiguous)
  v16h qa[4];
  {
    const float* qrow =
        q + (((size_t)b * S + (m0 + ln)) * H2 + (2 * h + grp)) * HD;
    #pragma unroll
    for (int f = 0; f < 4; ++f) {
      const int base0 = 32 * f + 16 * hl;
      v16h t;
      #pragma unroll
      for (int i = 0; i < 8; ++i) t[i]     = (_Float16)(qrow[base0 + i] * scale);
      #pragma unroll
      for (int i = 0; i < 8; ++i) t[8 + i] = (_Float16)(qrow[base0 + 8 + i] * scale);
      qa[f] = t;
    }
  }

  // accumulators: 16 tiles (16x16 f32, C layout) = rows m0..m0+15 x 256 vdims
  v8f o[16];
  {
    v8f z = {};
    #pragma unroll
    for (int i = 0; i < 16; ++i) o[i] = z;
  }
  float mrow[8], lrow[8];
  #pragma unroll
  for (int j = 0; j < 8; ++j) { mrow[j] = -__builtin_inff(); lrow[j] = 0.f; }

  const int nj = mb / BN + 1;  // key blocks up to and including the diagonal
  for (int jb = 0; jb < nj; ++jb) {
    const int j0 = jb * BN;
    __syncthreads();  // previous iteration's LDS reads complete

    // ---- cooperative load K (both groups), f32 -> f16, [g][key][d]
    // float4-chunked with limited unroll to keep staging registers small.
    {
      int r = tid >> 1, g = r >> 5, key = r & 31, d0 = (tid & 1) * 64;
      const float4* src = (const float4*)(
          k + (((size_t)b * S + (j0 + key)) * H2 + (2 * h + g)) * HD + d0);
      v8h* dst = (v8h*)(k_lds + (g * 32 + key) * 128 + d0);
      #pragma unroll 2
      for (int i = 0; i < 8; ++i) {
        float4 a = src[2 * i], c = src[2 * i + 1];
        v8h t;
        t[0] = (_Float16)a.x; t[1] = (_Float16)a.y;
        t[2] = (_Float16)a.z; t[3] = (_Float16)a.w;
        t[4] = (_Float16)c.x; t[5] = (_Float16)c.y;
        t[6] = (_Float16)c.z; t[7] = (_Float16)c.w;
        dst[i] = t;
      }
    }
    // ---- cooperative load V transposed: v_lds[vd][key], key-pairs packed b32.
    // thread -> (key pair 2kp/2kp+1, 32-vdim chunk); vd>=128 -> head 2h+1
    {
      int kp = tid & 15, vd0 = (tid >> 4) * 32;
      const float* s0 = v +
          (((size_t)b * S + (j0 + 2 * kp)) * H2 + 2 * h + (vd0 >> 7)) * HD +
          (vd0 & 127);
      const float* s1 = s0 + H2 * HD;  // next key, same head
      #pragma unroll 4
      for (int i = 0; i < 32; ++i) {
        v2h t;
        t[0] = (_Float16)s0[i];
        t[1] = (_Float16)s1[i];
        *(v2h*)(v_lds + (vd0 + i) * 32 + 2 * kp) = t;
      }
    }
    __syncthreads();

    // ---- scores: two 16x16 tiles (keys j0..j0+15, j0+16..j0+31)
    v8f c0 = {}, c1 = {};
    const _Float16* kg = k_lds + grp * (32 * 128);
    #pragma unroll
    for (int f = 0; f < 4; ++f) {
      // B layout: lane col n = ln; halves i: Kdim = 32f + 16*hl + i (contiguous)
      v16h b0 = *(const v16h*)(kg + ln * 128 + 32 * f + 16 * hl);
      v16h b1 = *(const v16h*)(kg + (16 + ln) * 128 + 32 * f + 16 * hl);
      c0 = __builtin_amdgcn_wmma_f32_16x16x32_f16(false, qa[f], false, b0,
                                                  (short)0, c0, false, false);
      c1 = __builtin_amdgcn_wmma_f32_16x16x32_f16(false, qa[f], false, b1,
                                                  (short)0, c1, false, false);
    }

    // ---- causal mask + online softmax (row m = m0 + j + 8*hl, col = ln),
    // P stored to LDS staging as it is produced (no p0/p1 register arrays).
    _Float16* pw = p_lds + wid * 512;
    float al[8];
    #pragma unroll
    for (int j = 0; j < 8; ++j) {
      const int m = m0 + j + 8 * hl;
      const int mr = j + 8 * hl;
      float s0 = (j0 + ln      <= m) ? c0[j] : -__builtin_inff();
      float s1 = (j0 + 16 + ln <= m) ? c1[j] : -__builtin_inff();
      const float t = bfly16_max(fmaxf(s0, s1));
      const float mnew = fmaxf(mrow[j], t);
      const float a = __expf(mrow[j] - mnew);
      mrow[j] = mnew;
      al[j] = a;
      const float e0 = __expf(s0 - mnew);
      const float e1 = __expf(s1 - mnew);
      pw[mr * 32 + ln]      = (_Float16)e0;
      pw[mr * 32 + 16 + ln] = (_Float16)e1;
      lrow[j] = lrow[j] * a + bfly16_sum(e0 + e1);
    }

    // rescale accumulators
    #pragma unroll
    for (int nt = 0; nt < 16; ++nt)
      #pragma unroll
      for (int j = 0; j < 8; ++j) o[nt][j] *= al[j];

    // ---- P: read back in A layout (same-wave DS ordering, no barrier needed)
    v16h pa;
    {
      v8h lo = *(const v8h*)(pw + ln * 32 + 16 * hl);
      v8h hi = *(const v8h*)(pw + ln * 32 + 8 + 16 * hl);
      #pragma unroll
      for (int i = 0; i < 8; ++i) { pa[i] = lo[i]; pa[8 + i] = hi[i]; }
    }

    // ---- P(16x32) x V(32x256): 16 WMMAs
    #pragma unroll
    for (int nt = 0; nt < 16; ++nt) {
      v16h bv = *(const v16h*)(v_lds + (16 * nt + ln) * 32 + 16 * hl);
      o[nt] = __builtin_amdgcn_wmma_f32_16x16x32_f16(false, pa, false, bv,
                                                     (short)0, o[nt], false, false);
    }
  }

  __syncthreads();  // done with k/v/p LDS regions
  const float lamf = s_lambda;

  float rinv[8];
  #pragma unroll
  for (int j = 0; j < 8; ++j) rinv[j] = 1.0f / lrow[j];

  if (grp == 1) {  // attn2 waves stage normalized output
    float* dst = cb + slc * (16 * 256);
    #pragma unroll
    for (int nt = 0; nt < 16; ++nt)
      #pragma unroll
      for (int j = 0; j < 8; ++j)
        dst[(j + 8 * hl) * 256 + 16 * nt + ln] = o[nt][j] * rinv[j];
  }
  __syncthreads();

  if (grp == 0) {  // combine, RMS-norm over 256 dims, write out
    const float* src = cb + slc * (16 * 256);
    float ss[8] = {0, 0, 0, 0, 0, 0, 0, 0};
    #pragma unroll
    for (int nt = 0; nt < 16; ++nt)
      #pragma unroll
      for (int j = 0; j < 8; ++j) {
        float x = o[nt][j] * rinv[j] - lamf * src[(j + 8 * hl) * 256 + 16 * nt + ln];
        o[nt][j] = x;
        ss[j] += x * x;
      }
    #pragma unroll
    for (int j = 0; j < 8; ++j)
      ss[j] = rsqrtf(bfly16_sum(ss[j]) * (1.0f / 256.0f) + 1e-5f) *
              (1.0f - LAM_INIT);
    #pragma unroll
    for (int nt = 0; nt < 16; ++nt) {
      const int head = 2 * h + (nt >> 3);      // '(H two) D' interleave
      const int dcol = 16 * (nt & 7) + ln;
      #pragma unroll
      for (int j = 0; j < 8; ++j) {
        const int si = m0 + j + 8 * hl;
        out[(((size_t)b * S + si) * H2 + head) * HD + dcol] = o[nt][j] * ss[j];
      }
    }
  }
}

extern "C" void kernel_launch(void* const* d_in, const int* in_sizes, int n_in,
                              void* d_out, int out_size, void* d_ws, size_t ws_size,
                              hipStream_t stream) {
  const float* q   = (const float*)d_in[0];
  const float* k   = (const float*)d_in[1];
  const float* v   = (const float*)d_in[2];
  const float* lq1 = (const float*)d_in[3];
  const float* lk1 = (const float*)d_in[4];
  const float* lq2 = (const float*)d_in[5];
  const float* lk2 = (const float*)d_in[6];
  float* out = (float*)d_out;

  const int B = 2;
  const int S = in_sizes[0] / (B * H2 * HD);  // 2048

  dim3 grid(S / BM, NH, B);
  dim3 block(128);
  FlashDiffAttention_82789789598491_kernel<<<grid, block, 0, stream>>>(
      q, k, v, lq1, lk1, lq2, lk2, out, S);
}
